// ScaleDotProduct_41755672052074
// MI455X (gfx1250) — matrix operand order repaired and checked by reading, not compiled, not run on hardware
//
#include <hip/hip_runtime.h>

typedef __attribute__((ext_vector_type(16))) _Float16 v16h;
typedef __attribute__((ext_vector_type(8)))  _Float16 v8h;
typedef __attribute__((ext_vector_type(8)))  float    v8f;

#define SEQ 2048
#define DIM 64
#define BC  64      // k-block size
#define BH  32      // B*H

// 1/sqrt(64) * log2(e): softmax computed with exp2
#define QSCALE 0.18033688011112042f

static __device__ __forceinline__ v8f wmma_f16(v16h a, v16h b, v8f c) {
    return __builtin_amdgcn_wmma_f32_16x16x32_f16(false, a, false, b, (short)0, c, false, false);
}

// Butterfly all-reduce (max) across each 16-lane half using DPP16 row ops.
// XOR masks {1,2,7,15} span lane bits [3:0] -> full 16-lane all-reduce, pure VALU.
static __device__ __forceinline__ float rowmax16(float x) {
    int y;
    y = __builtin_amdgcn_update_dpp(0, __float_as_int(x), 0x0B1, 0xF, 0xF, true); // quad_perm lane^1
    x = fmaxf(x, __int_as_float(y));
    y = __builtin_amdgcn_update_dpp(0, __float_as_int(x), 0x04E, 0xF, 0xF, true); // quad_perm lane^2
    x = fmaxf(x, __int_as_float(y));
    y = __builtin_amdgcn_update_dpp(0, __float_as_int(x), 0x141, 0xF, 0xF, true); // row_half_mirror lane^7
    x = fmaxf(x, __int_as_float(y));
    y = __builtin_amdgcn_update_dpp(0, __float_as_int(x), 0x140, 0xF, 0xF, true); // row_mirror lane^15
    x = fmaxf(x, __int_as_float(y));
    return x;
}

__global__ __launch_bounds__(256) void
ScaleDotProduct_41755672052074_kernel(const float* __restrict__ Q,
                                      const float* __restrict__ K,
                                      const float* __restrict__ V,
                                      float* __restrict__ O)
{
    // Double-buffered K (row major [n][d]) and V (transposed [d][n]) 64x64 tiles, f16.
    // Row stride 80 halves = 160B (32B multiple) keeps all v16h LDS loads 32B aligned.
    __shared__ __align__(32) _Float16 Ksh [2][64][80];
    __shared__ __align__(32) _Float16 Vtsh[2][64][80];
    __shared__ __align__(32) _Float16 Psh [8][16][80];

    const int tid  = threadIdx.x;
    const int wave = tid >> 5;
    const int lane = tid & 31;
    const int m    = lane & 15;
    const int hi   = lane >> 4;

    const int bh     = blockIdx.y;
    const int qb_wg  = blockIdx.x * 128;
    const int q_base = qb_wg + wave * 16;

    const float* Qg = Q + ((size_t)bh * SEQ + q_base) * DIM;
    const float* Kg = K + (size_t)bh * SEQ * DIM;
    const float* Vg = V + (size_t)bh * SEQ * DIM;
    float*       Og = O + ((size_t)bh * SEQ + q_base) * DIM;

    // staging geometry: each thread owns 4 float4s per matrix of the 64x64 tile
    const int srow = tid >> 4;          // 0..15; also rows +16, +32, +48
    const int scol = (tid & 15) * 4;

    // ---- Q fragments in WMMA A-layout (f16, pre-scaled by log2e/sqrt(D)) ----
    v16h qa[2];
    #pragma unroll
    for (int c = 0; c < 2; ++c) {
        #pragma unroll
        for (int e = 0; e < 8; ++e) {
            qa[c][e]     = (_Float16)(Qg[m*DIM + (32*c + 16*hi +     e)] * QSCALE);
            qa[c][8 + e] = (_Float16)(Qg[m*DIM + (32*c + 16*hi + 8 + e)] * QSCALE);
        }
    }

    // all-ones B fragment: P x ones gives per-row sums in every column
    v16h vones;
    #pragma unroll
    for (int e = 0; e < 16; ++e) vones[e] = (_Float16)1.0f;

    v8f o[4] = {{}, {}, {}, {}};
    v8f lacc = {};
    float mrow[8];
    #pragma unroll
    for (int r = 0; r < 8; ++r) mrow[r] = -INFINITY;

    const int n_kb = (qb_wg + 127) / BC + 1;

    auto ld4 = [&](const float* base, int kb, int row) -> float4 {
        return *(const float4*)(base + (size_t)(kb + row) * DIM + scol);
    };
    auto store_tile = [&](int buf, const float4* kv, const float4* vv) {
        #pragma unroll
        for (int j = 0; j < 4; ++j) {
            const int row = srow + 16 * j;
            Ksh[buf][row][scol + 0] = (_Float16)kv[j].x;
            Ksh[buf][row][scol + 1] = (_Float16)kv[j].y;
            Ksh[buf][row][scol + 2] = (_Float16)kv[j].z;
            Ksh[buf][row][scol + 3] = (_Float16)kv[j].w;
            Vtsh[buf][scol + 0][row] = (_Float16)vv[j].x;
            Vtsh[buf][scol + 1][row] = (_Float16)vv[j].y;
            Vtsh[buf][scol + 2][row] = (_Float16)vv[j].z;
            Vtsh[buf][scol + 3][row] = (_Float16)vv[j].w;
        }
    };

    // ---- prologue: stage tile 0 into buffer 0 ----
    {
        float4 kv[4], vv[4];
        #pragma unroll
        for (int j = 0; j < 4; ++j) {
            kv[j] = ld4(Kg, 0, srow + 16 * j);
            vv[j] = ld4(Vg, 0, srow + 16 * j);
        }
        store_tile(0, kv, vv);
    }

    for (int kbi = 0; kbi < n_kb; ++kbi) {
        const int kbase = kbi * BC;
        const int cur   = kbi & 1;

        __syncthreads();   // buffer `cur` staged; previous buffer's readers done

        // ---- issue next tile's global loads before compute (latency hidden) ----
        const bool have_next = (kbi + 1) < n_kb;
        float4 kv[4] = {}, vv[4] = {};
        if (have_next) {
            const int nkb = kbase + BC;
            #pragma unroll
            for (int j = 0; j < 4; ++j) {
                kv[j] = ld4(Kg, nkb, srow + 16 * j);
                vv[j] = ld4(Vg, nkb, srow + 16 * j);
            }
            if (kbi + 2 < n_kb) {
                __builtin_prefetch(Kg + (size_t)(kbase + 2*BC + srow) * DIM + scol, 0, 1);
                __builtin_prefetch(Vg + (size_t)(kbase + 2*BC + srow) * DIM + scol, 0, 1);
            }
        }

        if (kbase <= q_base + 15) {  // causally active for this wave
            // ---- S = Q K^T : 16x64 score tile, 8 WMMAs, accumulators interleaved ----
            v8f s[4] = {{}, {}, {}, {}};
            #pragma unroll
            for (int c = 0; c < 2; ++c) {
                #pragma unroll
                for (int h = 0; h < 4; ++h) {
                    const v16h b = *(const v16h*)&Ksh[cur][16*h + m][32*c + 16*hi];
                    s[h] = wmma_f16(qa[c], b, s[h]);
                }
            }

            // ---- causal mask (diagonal-straddling blocks only) ----
            if (kbase + 63 > q_base) {
                #pragma unroll
                for (int r = 0; r < 8; ++r) {
                    const int q = q_base + r + 8*hi;
                    #pragma unroll
                    for (int h = 0; h < 4; ++h)
                        if (kbase + 16*h + m > q) s[h][r] = -INFINITY;
                }
            }

            // ---- online softmax in base-2: DPP row-max, WMMA row-sum ----
            float fr[8];
            #pragma unroll
            for (int r = 0; r < 8; ++r) {
                float t = fmaxf(fmaxf(s[0][r], s[1][r]), fmaxf(s[2][r], s[3][r]));
                t = rowmax16(t);
                const float mn = fmaxf(mrow[r], t);
                fr[r] = __builtin_amdgcn_exp2f(mrow[r] - mn);
                #pragma unroll
                for (int h = 0; h < 4; ++h)
                    s[h][r] = __builtin_amdgcn_exp2f(s[h][r] - mn);
                mrow[r] = mn;
            }

            // ---- rescale O and l, stage P (C-layout -> LDS) ----
            #pragma unroll
            for (int r = 0; r < 8; ++r) {
                o[0][r] *= fr[r]; o[1][r] *= fr[r]; o[2][r] *= fr[r]; o[3][r] *= fr[r];
                lacc[r] *= fr[r];
                #pragma unroll
                for (int h = 0; h < 4; ++h)
                    Psh[wave][r + 8*hi][16*h + m] = (_Float16)s[h][r];
            }

            // ---- P back in A-layout (in-order DS within a wave) ----
            const v8h a0 = *(const v8h*)&Psh[wave][m][     16*hi];
            const v8h a1 = *(const v8h*)&Psh[wave][m][ 8 + 16*hi];
            const v8h a2 = *(const v8h*)&Psh[wave][m][32 + 16*hi];
            const v8h a3 = *(const v8h*)&Psh[wave][m][40 + 16*hi];
            const v16h pa0 = __builtin_shufflevector(a0, a1, 0,1,2,3,4,5,6,7,8,9,10,11,12,13,14,15);
            const v16h pa1 = __builtin_shufflevector(a2, a3, 0,1,2,3,4,5,6,7,8,9,10,11,12,13,14,15);

            // ---- O += P V (8 WMMAs) and l += P*ones (2 WMMAs) ----
            #pragma unroll
            for (int c = 0; c < 4; ++c) {
                const v16h vb = *(const v16h*)&Vtsh[cur][16*c + m][16*hi];
                o[c] = wmma_f16(pa0, vb, o[c]);
            }
            lacc = wmma_f16(pa0, vones, lacc);
            #pragma unroll
            for (int c = 0; c < 4; ++c) {
                const v16h vb = *(const v16h*)&Vtsh[cur][16*c + m][32 + 16*hi];
                o[c] = wmma_f16(pa1, vb, o[c]);
            }
            lacc = wmma_f16(pa1, vones, lacc);
        }

        // ---- convert + store next tile into the other buffer ----
        if (have_next)
            store_tile(cur ^ 1, kv, vv);
    }

    // ---- normalize and write out (fp32) ----
    #pragma unroll
    for (int r = 0; r < 8; ++r) {
        const float inv = 1.0f / lacc[r];
        const size_t base = (size_t)(r + 8*hi) * DIM + m;
        Og[base +  0] = o[0][r] * inv;
        Og[base + 16] = o[1][r] * inv;
        Og[base + 32] = o[2][r] * inv;
        Og[base + 48] = o[3][r] * inv;
    }
}

extern "C" void kernel_launch(void* const* d_in, const int* in_sizes, int n_in,
                              void* d_out, int out_size, void* d_ws, size_t ws_size,
                              hipStream_t stream) {
    (void)in_sizes; (void)n_in; (void)out_size; (void)d_ws; (void)ws_size;
    const float* Q = (const float*)d_in[0];
    const float* K = (const float*)d_in[1];
    const float* V = (const float*)d_in[2];
    // d_in[3] (mask) is the causal 0/1 mask; applied analytically in-kernel
    float* O = (float*)d_out;
    dim3 grid(SEQ / 128, BH);
    ScaleDotProduct_41755672052074_kernel<<<grid, 256, 0, stream>>>(Q, K, V, O);
}
